// GraphUpdateLayer_81741817578211
// MI455X (gfx1250) — matrix operand order repaired and checked by reading, not compiled, not run on hardware
//
#include <hip/hip_runtime.h>
#include <hip/hip_bf16.h>

// ---------------------------------------------------------------------------
// MI455X (gfx1250) implementation.
// Both convs run as implicit GEMM on v_wmma_f32_16x16x32_f16 (wave32):
//   A (16x32 f16)  = packed weight fragments, pre-laid-out in contiguous k order (k = 16*(lane/16) + s)
//   B (32x16 f16)  = im2col slice read from an LDS tile (channel-last)
//   C/D (16x16 f32)= 8-VGPR accumulators per M-tile
// Stage-2 staging uses GLOBAL_LOAD_ASYNC_TO_LDS_B128 (ASYNCcnt) with double-
// buffered neighbor tiles so DMA overlaps WMMA compute.
// ---------------------------------------------------------------------------

typedef __attribute__((ext_vector_type(16))) _Float16 v16h;
typedef __attribute__((ext_vector_type(8)))  float    v8f;
typedef __attribute__((ext_vector_type(4)))  unsigned int u32x4;

union V16U { v16h h; u32x4 u[2]; };

#define HW  96
#define NND 32      // nodes
#define EPS 1e-5f

#if defined(__has_builtin)
#if __has_builtin(__builtin_amdgcn_global_load_async_to_lds_b128) && \
    __has_builtin(__builtin_amdgcn_s_wait_asynccnt)
#define HAS_ASYNC 1
#else
#define HAS_ASYNC 0
#endif
#else
#define HAS_ASYNC 0
#endif

#if HAS_ASYNC
// Param types per hipcc diagnostic: 'int __vector(4) __device__ *' (AS1) for the
// global source; symmetric AS3 vector pointer for the LDS destination.
typedef int i32x4_vs __attribute__((vector_size(16)));
typedef __attribute__((address_space(1))) i32x4_vs* async_gptr;
typedef __attribute__((address_space(3))) i32x4_vs* async_lptr;
#endif

__device__ __forceinline__ float elu(float v) {
    return v > 0.f ? v : __expf(v) - 1.f;
}

// ---------------------------------------------------------------------------
// X (n,c,y,x) f32  ->  Xc (n,y,x,c) f16   (channel-last for contiguous LDS fill)
// total = 32*96*96*32 threads
__global__ void k_castX(const float* __restrict__ X, _Float16* __restrict__ Xc) {
    int idx = blockIdx.x * 256 + threadIdx.x;
    int c = idx & 31;
    int pix = idx >> 5;
    int x = pix % HW;
    int t = pix / HW;
    int y = t % HW;
    int n = t / HW;
    Xc[(size_t)pix * 32 + c] =
        (_Float16)X[(((size_t)n * 32 + c) * HW + y) * HW + x];
}

// ---------------------------------------------------------------------------
// Pack W0 (n,co,ci,3,3) f32 into A-matrix fragments (16x32 A, contiguous k order):
//   lane l<16 : M=l,   K = {0..15}
//   lane l>=16: M=l-16,K = {16..31}
// frag index = (n*2 + mtile)*9 + tap ; 512 halves per frag, lane-major.
__global__ void k_packW0(const float* __restrict__ W0, _Float16* __restrict__ pW0) {
    int idx = blockIdx.x * 256 + threadIdx.x;
    int j = idx & 15;
    int l = (idx >> 4) & 31;
    int frag = idx >> 9;
    int t = frag % 9;
    int rem = frag / 9;
    int m = rem & 1;
    int n = rem >> 1;
    int kk = j + 16 * (l >> 4);   // K index 0..31 (= ci)
    int co = m * 16 + (l & 15);
    int r = t / 3, s = t % 3;
    pW0[idx] = (_Float16)W0[((((size_t)n * 32 + co) * 32 + kk) * 3 + r) * 3 + s];
}

// Pack W1 (n,k,co,ci(64),3,3) f32, chunk = tap*2 + ci_half (18 chunks of K=32).
// frag index = ((n*4+k)*2 + mtile)*18 + chunk
__global__ void k_packW1(const float* __restrict__ W1, _Float16* __restrict__ pW1) {
    int idx = blockIdx.x * 256 + threadIdx.x;
    int j = idx & 15;
    int l = (idx >> 4) & 31;
    int frag = idx >> 9;
    int c = frag % 18;
    int rem = frag / 18;
    int m = rem & 1;
    int nk = rem >> 1;                              // n*4 + k
    int t = c >> 1, h = c & 1;
    int kk = j + 16 * (l >> 4);
    int ci = h * 32 + kk;
    int co = m * 16 + (l & 15);
    int r = t / 3, s = t % 3;
    pW1[idx] = (_Float16)W1[(((size_t)nk * 32 + co) * 64 + ci) * 9 + r * 3 + s];
}

// ---------------------------------------------------------------------------
// Stage 1: per-node conv3x3 (32->32) + bias + ELU. Writes un-normalized result
// into Xtmp channels [0,32) (channel-last f16) and per-(n,co,y,wave) partial
// sum / sumsq for the instance norm (deterministic two-pass reduction).
// Block = (y, n); 192 threads = 6 waves; wave w owns pixels x in [16w,16w+16).
__global__ void __launch_bounds__(192)
k_stage1(const _Float16* __restrict__ Xc, const _Float16* __restrict__ pW0,
         const float* __restrict__ b0, _Float16* __restrict__ Xtmp,
         float* __restrict__ psum, float* __restrict__ psumsq) {
    const int y = blockIdx.x, n = blockIdx.y;
    const int tid = threadIdx.x;
    const int w = tid >> 5, l = tid & 31;
    const int hi = l >> 4, ln = l & 15;

    // LDS tile: rows y-1..y+1, cols -1..96, 32 ch. pitch 40 halves (80B, 16B-mult)
    __shared__ __align__(16) _Float16 tile[3 * 98 * 40];

    for (int q = tid; q < 1176; q += 192) {            // 3*98 positions * 4 x b128
        int part = q & 3;
        int rc = q >> 2;
        int row = rc / 98, col = rc % 98;
        int gy = y + row - 1, gx = col - 1;
        u32x4 v = {};
        if ((unsigned)gy < (unsigned)HW && (unsigned)gx < (unsigned)HW)
            v = *(const u32x4*)(Xc + ((((size_t)n * HW + gy) * HW + gx) * 32 + part * 8));
        *(u32x4*)(tile + (rc * 40 + part * 8)) = v;
    }
    __syncthreads();

    v8f acc0 = {}, acc1 = {};
    const _Float16* pA = pW0 + (size_t)n * 2 * 9 * 512;
    for (int t = 0; t < 9; ++t) {
        int r = t / 3, s = t % 3;
        v16h a0 = *(const v16h*)(pA + (size_t)(0 * 9 + t) * 512 + l * 16);
        v16h a1 = *(const v16h*)(pA + (size_t)(1 * 9 + t) * 512 + l * 16);
        int col = 16 * w + ln + s;                     // lds col (halo built in)
        V16U b;
        const u32x4* bp = (const u32x4*)(tile + ((r * 98 + col) * 40 + hi * 16));
        b.u[0] = bp[0];
        b.u[1] = bp[1];
        acc0 = __builtin_amdgcn_wmma_f32_16x16x32_f16(false, a0, false, b.h,
                                                      (short)0, acc0, false, false);
        acc1 = __builtin_amdgcn_wmma_f32_16x16x32_f16(false, a1, false, b.h,
                                                      (short)0, acc1, false, false);
    }

    const int x = 16 * w + ln;
    for (int m = 0; m < 2; ++m) {
        v8f acc = m ? acc1 : acc0;
#pragma unroll
        for (int jv = 0; jv < 8; ++jv) {
            int co = m * 16 + jv + 8 * hi;
            float v = acc[jv] + b0[n * 32 + co];
            v = elu(v);
            Xtmp[(((size_t)n * HW + y) * HW + x) * 64 + co] = (_Float16)v;
            // reduce over the 16 pixels of this N-tile (lanes with same hi)
            float s1 = v, s2 = v * v;
#pragma unroll
            for (int off = 8; off >= 1; off >>= 1) {
                s1 += __shfl_xor(s1, off, 16);
                s2 += __shfl_xor(s2, off, 16);
            }
            if (ln == 0) {
                size_t o = (((size_t)n * 32 + co) * HW + y) * 6 + w;
                psum[o] = s1;
                psumsq[o] = s2;
            }
        }
    }
}

// ---------------------------------------------------------------------------
// Reduce 96*6 partials per (n,c) -> mu, rstd (fixed order => deterministic).
__global__ void k_stats(const float* __restrict__ psum, const float* __restrict__ psumsq,
                        float* __restrict__ mu, float* __restrict__ rstd) {
    int idx = blockIdx.x * 256 + threadIdx.x;
    if (idx >= NND * 32) return;
    const float* ps = psum + (size_t)idx * 576;
    const float* pq = psumsq + (size_t)idx * 576;
    float s = 0.f, q = 0.f;
    for (int i = 0; i < 576; ++i) { s += ps[i]; q += pq[i]; }
    const float inv = 1.f / (float)(HW * HW);
    float m = s * inv;
    float v = q * inv - m * m;
    v = v < 0.f ? 0.f : v;
    mu[idx] = m;
    rstd[idx] = rsqrtf(v + EPS);
}

// ---------------------------------------------------------------------------
// Normalize Xtmp channels [0,32) in place, fill channels [32,64) from Xhid.
__global__ void k_normcat(const float* __restrict__ Xhid, const float* __restrict__ mu,
                          const float* __restrict__ rstd, _Float16* __restrict__ Xtmp) {
    size_t idx = (size_t)blockIdx.x * 256 + threadIdx.x;
    int c = (int)(idx & 63);
    size_t pix = idx >> 6;
    int x = (int)(pix % HW);
    size_t t = pix / HW;
    int y = (int)(t % HW);
    int n = (int)(t / HW);
    if (c < 32) {
        float v = (float)Xtmp[pix * 64 + c];
        Xtmp[pix * 64 + c] = (_Float16)((v - mu[n * 32 + c]) * rstd[n * 32 + c]);
    } else {
        Xtmp[pix * 64 + c] =
            (_Float16)Xhid[(((size_t)n * 32 + (c - 32)) * HW + y) * HW + x];
    }
}

// ---------------------------------------------------------------------------
// Stage 2: for node n, row y: loop 4 neighbors with DOUBLE-BUFFERED async
// LDS staging (3 x 98 x 64ch f16 per neighbor); 18 K-chunks x 2 M-tiles of
// wmma per wave; mean over k, +bias, ELU, write f32 NCHW output.
__global__ void __launch_bounds__(192)
k_stage2(const _Float16* __restrict__ Xtmp, const _Float16* __restrict__ pW1,
         const float* __restrict__ b1, const int* __restrict__ nn_idx,
         float* __restrict__ out) {
    const int y = blockIdx.x, n = blockIdx.y;
    const int tid = threadIdx.x;
    const int w = tid >> 5, l = tid & 31;
    const int hi = l >> 4, ln = l & 15;

    // pitch 72 halves = 144B per (row,col) -> every b128 slot is 16B aligned
    __shared__ __align__(16) _Float16 tile[2][3 * 98 * 72];

    // Zero both buffers once: the SAME-padding halo stays zero; interior is
    // overwritten by staging each iteration.
    for (int q = tid; q < 2 * 2646; q += 192)          // 2*3*98*72*2B / 16B
        ((u32x4*)tile)[q] = (u32x4){};

    // Stage one neighbor tile into tile[buf] (interior positions only).
    auto stage = [&](int buf, int nbr) {
        for (int q = tid; q < 2352; q += 192) {        // 3*98 positions * 8 x b128
            int part = q & 7;
            int rc = q >> 3;
            int row = rc / 98, col = rc % 98;
            int gy = y + row - 1, gx = col - 1;
            if ((unsigned)gy < (unsigned)HW && (unsigned)gx < (unsigned)HW) {
                const _Float16* src = Xtmp +
                    ((((size_t)nbr * HW + gy) * HW + gx) * 64 + part * 8);
                _Float16* dst = &tile[buf][rc * 72 + part * 8];
#if HAS_ASYNC
                __builtin_amdgcn_global_load_async_to_lds_b128(
                    (async_gptr)src, (async_lptr)dst, 0, 0);
#else
                *(u32x4*)dst = *(const u32x4*)src;
#endif
            }
        }
    };

    __syncthreads();                 // zero-fill visible to all waves
    stage(0, nn_idx[n * 4 + 0]);     // prologue: prefetch neighbor 0

    v8f acc0 = {}, acc1 = {};
    for (int k = 0; k < 4; ++k) {
#if HAS_ASYNC
        __builtin_amdgcn_s_wait_asynccnt(0);   // my async stores to LDS done
#endif
        __syncthreads();                       // everyone's staging done
        if (k < 3) stage((k + 1) & 1, nn_idx[n * 4 + k + 1]);  // overlap DMA

        const _Float16* buf = tile[k & 1];
        const _Float16* pA = pW1 + ((size_t)(n * 4 + k) * 2) * 18 * 512;
#pragma unroll 2
        for (int c = 0; c < 18; ++c) {
            int t = c >> 1, h = c & 1;
            int r = t / 3, s = t - r * 3;
            v16h a0 = *(const v16h*)(pA + (size_t)(0 * 18 + c) * 512 + l * 16);
            v16h a1 = *(const v16h*)(pA + (size_t)(1 * 18 + c) * 512 + l * 16);
            int col = 16 * w + ln + s;
            V16U b;
            const u32x4* bp =
                (const u32x4*)(buf + ((r * 98 + col) * 72 + h * 32 + hi * 16));
            b.u[0] = bp[0];
            b.u[1] = bp[1];
            acc0 = __builtin_amdgcn_wmma_f32_16x16x32_f16(false, a0, false, b.h,
                                                          (short)0, acc0, false, false);
            acc1 = __builtin_amdgcn_wmma_f32_16x16x32_f16(false, a1, false, b.h,
                                                          (short)0, acc1, false, false);
        }
    }

    const int x = 16 * w + ln;
    for (int m = 0; m < 2; ++m) {
        v8f acc = m ? acc1 : acc0;
#pragma unroll
        for (int jv = 0; jv < 8; ++jv) {
            int co = m * 16 + jv + 8 * hi;
            float mb = 0.25f * (b1[(n * 4 + 0) * 32 + co] + b1[(n * 4 + 1) * 32 + co] +
                                b1[(n * 4 + 2) * 32 + co] + b1[(n * 4 + 3) * 32 + co]);
            float v = acc[jv] * 0.25f + mb;
            out[(((size_t)n * 32 + co) * HW + y) * HW + x] = elu(v);
        }
    }
}

// ---------------------------------------------------------------------------
extern "C" void kernel_launch(void* const* d_in, const int* in_sizes, int n_in,
                              void* d_out, int out_size, void* d_ws, size_t ws_size,
                              hipStream_t stream) {
    const float* X    = (const float*)d_in[0];
    const float* Xhid = (const float*)d_in[1];
    const float* W0   = (const float*)d_in[2];
    const float* b0   = (const float*)d_in[3];
    const float* W1   = (const float*)d_in[4];
    const float* b1   = (const float*)d_in[5];
    const int*   nn   = (const int*)d_in[6];
    float* out = (float*)d_out;

    char* ws = (char*)d_ws;
    size_t off = 0;
    auto arena = [&](size_t bytes) {
        size_t o = off;
        off += (bytes + 255) & ~(size_t)255;
        return o;
    };
    _Float16* Xc     = (_Float16*)(ws + arena((size_t)32 * 96 * 96 * 32 * 2)); // 18.9MB
    _Float16* Xtmp   = (_Float16*)(ws + arena((size_t)32 * 96 * 96 * 64 * 2)); // 37.7MB
    _Float16* pW0    = (_Float16*)(ws + arena((size_t)32 * 2 * 9 * 512 * 2));
    _Float16* pW1    = (_Float16*)(ws + arena((size_t)32 * 4 * 2 * 18 * 512 * 2));
    float*    psum   = (float*)(ws + arena((size_t)1024 * 96 * 6 * 4));
    float*    psumsq = (float*)(ws + arena((size_t)1024 * 96 * 6 * 4));
    float*    mu     = (float*)(ws + arena((size_t)1024 * 4));
    float*    rstd   = (float*)(ws + arena((size_t)1024 * 4));

    k_castX<<<36864, 256, 0, stream>>>(X, Xc);
    k_packW0<<<1152, 256, 0, stream>>>(W0, pW0);
    k_packW1<<<9216, 256, 0, stream>>>(W1, pW1);

    dim3 grid(96, 32);
    k_stage1<<<grid, 192, 0, stream>>>(Xc, pW0, b0, Xtmp, psum, psumsq);
    k_stats<<<4, 256, 0, stream>>>(psum, psumsq, mu, rstd);
    k_normcat<<<73728, 256, 0, stream>>>(Xhid, mu, rstd, Xtmp);
    k_stage2<<<grid, 192, 0, stream>>>(Xtmp, pW1, b1, nn, out);
}
